// Translate_19825569038896
// MI455X (gfx1250) — matrix operand order repaired and checked by reading, not compiled, not run on hardware
//
#include <hip/hip_runtime.h>

typedef float    v8f  __attribute__((ext_vector_type(8)));
typedef float    v4f  __attribute__((ext_vector_type(4)));
typedef _Float16 v16h __attribute__((ext_vector_type(16)));
typedef _Float16 v8h  __attribute__((ext_vector_type(8)));
typedef _Float16 v4h  __attribute__((ext_vector_type(4)));
typedef unsigned int uint32x4 __attribute__((ext_vector_type(4)));
typedef int          int32x8  __attribute__((ext_vector_type(8)));
typedef int          int32x4  __attribute__((ext_vector_type(4)));

union AFrag { v16h v; v8h h[2]; };

#define N_ROWS   65536
#define M_TILE   32
#define NX       1024
#define NH       512

// ---- LDS layout (dynamic, byte offsets from base of dynamic LDS) ----
#define OFF_ACC  0u                       // 128 floats (l1,l2,l3,l4 x 32)
#define OFF_A    512u                     // region A: 64 KB (sX; GEMM2 W buf0)
#define OFF_B    66048u                   // region B: 64 KB (GEMM1 W dbl-buf; GEMM2 W buf1)
#define OFF_H1   131584u                  // 32 KB  H1 f16
#define OFF_H2   164352u                  // 32 KB  H2 f16
#define SMEM_SZ  197120u

__device__ inline v8f zero8() {
  v8f z;
#pragma unroll
  for (int i = 0; i < 8; ++i) z[i] = 0.f;
  return z;
}

// A fragment (16x32 f16, M x K) from row-major f16 tile in LDS.
// lanes 0-15: M=lane, K {k0..k0+7, k0+16..k0+23}; lanes 16-31: M=lane-16, K {+8..+15,+24..+31}.
__device__ inline v16h load_a_frag(const _Float16* base, int pitch, int m0, int k0, int lane) {
  int r  = m0 + (lane & 15);
  int kb = k0 + ((lane >> 4) << 3);
  const _Float16* p = base + r * pitch + kb;
  AFrag f;
  f.h[0] = *(const v8h*)(p);
  f.h[1] = *(const v8h*)(p + 16);
  return f.v;
}

// B fragment (32x16 f16, K x N) from a TDM-staged LDS slice, pitch 32 halves per row n.
// lane n%16 -> column n0+n; lanes 0-15 K {0..7,16..23}, lanes 16-31 K {8..15,24..31} of the slice.
__device__ inline v16h load_b_lds(const _Float16* sW, int n0, int lane) {
  const _Float16* p = sW + (n0 + (lane & 15)) * 32 + ((lane >> 4) << 3);
  AFrag f;
  f.h[0] = *(const v8h*)(p);
  f.h[1] = *(const v8h*)(p + 16);
  return f.v;
}

// TDM: async 2-D f16 tile load Global->LDS (D# per CDNA5 ISA sec. 8.3/8.4).
// tile = tile_d0 (halves, contiguous) x tile_d1 (rows, stride d0_stride halves), packed into LDS.
// 6-arg builtin form (clang-23 / therock-10.0): (g0, g1, g2, g3, g4, cpol).
__device__ inline void tdm_load_2d_f16(unsigned lds_off, const _Float16* gptr,
                                       unsigned tensor_d0, unsigned tensor_d1,
                                       unsigned tile_d0, unsigned tile_d1,
                                       unsigned d0_stride) {
  unsigned long long ga = (unsigned long long)(size_t)gptr;
  uint32x4 g0;
  g0[0] = 1u;                                            // count=1, user descriptor
  g0[1] = lds_off;                                       // lds_addr (bytes)
  g0[2] = (unsigned)(ga & 0xFFFFFFFFu);                  // global_addr[31:0]
  g0[3] = (unsigned)((ga >> 32) & 0x01FFFFFFu) | (2u << 30);  // addr[56:32] | type=2
  int32x8 g1;
  g1[0] = (int)(1u << 16);                               // wg_mask=0, data_size=1 (2B)
  g1[1] = (int)((tensor_d0 & 0xFFFFu) << 16);            // tensor_dim0[15:0] @ [63:48]
  g1[2] = (int)(((tensor_d0 >> 16) & 0xFFFFu) | ((tensor_d1 & 0xFFFFu) << 16));
  g1[3] = (int)(((tensor_d1 >> 16) & 0xFFFFu) | (tile_d0 << 16));   // tile_dim0 @ [127:112]
  g1[4] = (int)(tile_d1 & 0xFFFFu);                      // tile_dim1; tile_dim2=0
  g1[5] = (int)d0_stride;                                // tensor_dim0_stride[31:0]
  g1[6] = 0;                                             // stride hi, dim1_stride lo
  g1[7] = 0;
  int32x4 z4 = {0, 0, 0, 0};
  int32x8 z8 = {0, 0, 0, 0, 0, 0, 0, 0};
  __builtin_amdgcn_tensor_load_to_lds(g0, g1, z4, z4, z8, 0);
}

__device__ inline void stage_x_f16(const float* __restrict__ Xg, _Float16* sX, int row0, int tid) {
  int r  = tid >> 3;             // 32 rows, 8 threads per row
  int cb = (tid & 7) * 128;
  const float* g = Xg + (size_t)(row0 + r) * NX + cb;
  _Float16* s = sX + r * NX + cb;
#pragma unroll 4
  for (int it = 0; it < 32; ++it) {
    v4f a = *(const v4f*)(g + it * 4);
    v4h ah;
#pragma unroll
    for (int e = 0; e < 4; ++e) ah[e] = (_Float16)a[e];
    *(v4h*)(s + it * 4) = ah;
  }
}

__global__ void cvt_f32_f16(const float* __restrict__ src, _Float16* __restrict__ dst, int n) {
  int i = blockIdx.x * blockDim.x + threadIdx.x;
  int stride = gridDim.x * blockDim.x;
  for (; i < n; i += stride) dst[i] = (_Float16)src[i];
}

__global__ void zero_f32(float* p, int n) {
  int i = blockIdx.x * blockDim.x + threadIdx.x;
  if (i < n) p[i] = 0.f;
}

__global__ __launch_bounds__(256) void fused_mlp_loss(
    const float* __restrict__ X1, const float* __restrict__ X2, const float* __restrict__ Y,
    const float* __restrict__ b1h, const float* __restrict__ b2h,
    const float* __restrict__ bh1, const float* __restrict__ bh2,
    const float* __restrict__ whoW, const float* __restrict__ whoB,
    const _Float16* __restrict__ w1h16, const _Float16* __restrict__ w2h16,
    const _Float16* __restrict__ wh1_16, const _Float16* __restrict__ wh2_16,
    float* __restrict__ out)
{
  extern __shared__ char smem_raw[];
  float* l1acc = (float*)(smem_raw + OFF_ACC);
  float* l2acc = l1acc + 32;
  float* l3acc = l1acc + 64;
  float* l4acc = l1acc + 96;
  _Float16* sX  = (_Float16*)(smem_raw + OFF_A);
  _Float16* sH1 = (_Float16*)(smem_raw + OFF_H1);
  _Float16* sH2 = (_Float16*)(smem_raw + OFF_H2);

  const int tid  = threadIdx.x;
  const int lane = tid & 31;
  const int wave = tid >> 5;
  const int row0 = blockIdx.x * M_TILE;
  const int mt   = wave & 1;       // M sub-tile (16 rows)
  const int ngrp = wave >> 1;      // 4 N-groups

  if (tid < 32) { l1acc[tid] = 0.f; l2acc[tid] = 0.f; l3acc[tid] = 0.f; l4acc[tid] = 0.f; }

  // ---- Phase 1: H1 = X1@w1h^T + b1h ; H2 = X2@w2h^T + b2h  (sequential X staging) ----
  for (int which = 0; which < 2; ++which) {
    const float*    Xg   = which ? X2    : X1;
    const _Float16* Wg   = which ? w2h16 : w1h16;
    const float*    bias = which ? b2h   : b1h;
    _Float16*       sh   = which ? sH2   : sH1;

    stage_x_f16(Xg, sX, row0, tid);
    if (wave == 0)   // prefetch weight slice 0 (512 rows x 32 halves = 32 KB)
      tdm_load_2d_f16(OFF_B, Wg, NX, NH, 32, NH, NX);

    v8f c[8];
#pragma unroll
    for (int j = 0; j < 8; ++j) c[j] = zero8();

    for (int i = 0; i < NX / 32; ++i) {
      const int k0 = i * 32;
      if (wave == 0) {
        if (i + 1 < NX / 32) {
          tdm_load_2d_f16(OFF_B + ((unsigned)(i + 1) & 1u) * 32768u,
                          Wg + (k0 + 32), NX - (k0 + 32), NH, 32, NH, NX);
          __builtin_amdgcn_s_wait_tensorcnt(1);   // slice i landed
        } else {
          __builtin_amdgcn_s_wait_tensorcnt(0);
        }
      }
      __syncthreads();   // slice i (and X staging at i==0) visible to all waves
      const _Float16* sW = (const _Float16*)(smem_raw + OFF_B + ((unsigned)i & 1u) * 32768u);
      v16h a = load_a_frag(sX, NX, mt * 16, k0, lane);
#pragma unroll
      for (int j = 0; j < 8; ++j) {
        v16h b = load_b_lds(sW, (ngrp * 8 + j) * 16, lane);
        c[j] = __builtin_amdgcn_wmma_f32_16x16x32_f16(false, a, false, b,
                                                      (short)0, c[j], false, false);
      }
      __syncthreads();   // all waves done with slice i before its buffer is re-filled
    }

    const int mbase = mt * 16 + ((lane >> 4) << 3);
#pragma unroll
    for (int j = 0; j < 8; ++j) {
      int n = (ngrp * 8 + j) * 16 + (lane & 15);
      float bv = bias[n];
#pragma unroll
      for (int r = 0; r < 8; ++r)
        sh[(mbase + r) * NH + n] = (_Float16)(c[j][r] + bv);
    }
  }
  __syncthreads();

  // ---- Phase 2: L3 = (H1-H2) row-sumsq ; L4 dot = (H1+H2) . who_w ----
  {
    int m  = tid >> 3;
    int cb = (tid & 7) * 64;
    const _Float16* h1p = sH1 + m * NH + cb;
    const _Float16* h2p = sH2 + m * NH + cb;
    const float*    wv  = whoW + cb;
    float p3 = 0.f, p4 = 0.f;
#pragma unroll 8
    for (int c = 0; c < 64; ++c) {
      float h1 = (float)h1p[c];
      float h2 = (float)h2p[c];
      float d  = h1 - h2;
      p3 += d * d;
      p4 += (h1 + h2) * wv[c];
    }
    atomicAdd(&l3acc[m], p3);
    atomicAdd(&l4acc[m], p4);
  }

  // ---- Phase 3: L1 = H1@wh2^T + bh2 - X2 ; L2 = H2@wh1^T + bh1 - X1 (row-sumsq) ----
  // W slices are 1024 rows x 32 halves = 64 KB, double-buffered in region A + region B.
  {
    const int mbase = mt * 16 + ((lane >> 4) << 3);
    for (int which = 0; which < 2; ++which) {
      const _Float16* Wg   = which ? wh1_16 : wh2_16;
      const _Float16* sh   = which ? sH2    : sH1;
      const float*    bias = which ? bh1    : bh2;
      const float*    Xg   = which ? X1     : X2;
      float*          lacc = which ? l2acc  : l1acc;
      for (int half = 0; half < 2; ++half) {
        const int nb = ngrp * 16 + half * 8;   // this wave's N-tile base (of 64)
        if (wave == 0)
          tdm_load_2d_f16(OFF_A, Wg, NH, NX, 32, NX, NH);   // slice 0 -> region A
        v8f c[8];
#pragma unroll
        for (int j = 0; j < 8; ++j) c[j] = zero8();

        for (int i = 0; i < NH / 32; ++i) {
          const int k0 = i * 32;
          if (wave == 0) {
            if (i + 1 < NH / 32) {
              tdm_load_2d_f16(((i + 1) & 1) ? OFF_B : OFF_A,
                              Wg + (k0 + 32), NH - (k0 + 32), NX, 32, NX, NH);
              __builtin_amdgcn_s_wait_tensorcnt(1);
            } else {
              __builtin_amdgcn_s_wait_tensorcnt(0);
            }
          }
          __syncthreads();
          const _Float16* sW = (const _Float16*)(smem_raw + ((i & 1) ? OFF_B : OFF_A));
          v16h a = load_a_frag(sh, NH, mt * 16, k0, lane);
#pragma unroll
          for (int j = 0; j < 8; ++j) {
            v16h b = load_b_lds(sW, (nb + j) * 16, lane);
            c[j] = __builtin_amdgcn_wmma_f32_16x16x32_f16(false, a, false, b,
                                                          (short)0, c[j], false, false);
          }
          __syncthreads();
        }

        float rs[8];
#pragma unroll
        for (int r = 0; r < 8; ++r) rs[r] = 0.f;
#pragma unroll
        for (int j = 0; j < 8; ++j) {
          int n = (nb + j) * 16 + (lane & 15);
          float bv = bias[n];
#pragma unroll
          for (int r = 0; r < 8; ++r) {
            float xv = Xg[(size_t)(row0 + mbase + r) * NX + n];
            float v  = c[j][r] + bv - xv;
            rs[r] += v * v;
          }
        }
#pragma unroll
        for (int r = 0; r < 8; ++r) {      // reduce across the 16 lanes of each row
          float v = rs[r];
          v += __shfl_xor(v, 1);
          v += __shfl_xor(v, 2);
          v += __shfl_xor(v, 4);
          v += __shfl_xor(v, 8);
          if ((lane & 15) == 0) atomicAdd(&lacc[mbase + r], v);
        }
      }
    }
  }
  __syncthreads();

  // ---- Phase 4: per-row combine (l1^2 + l2^2 + l3^2 + L4^2), block reduce, atomic add ----
  if (wave == 0) {
    float a1 = l1acc[lane], a2 = l2acc[lane], a3 = l3acc[lane];
    float L4 = (l4acc[lane] + whoB[0]) * 0.5f - Y[row0 + lane];
    float v  = a1 * a1 + a2 * a2 + a3 * a3 + L4 * L4;
    v += __shfl_xor(v, 16);
    v += __shfl_xor(v, 8);
    v += __shfl_xor(v, 4);
    v += __shfl_xor(v, 2);
    v += __shfl_xor(v, 1);
    if (lane == 0) atomicAdd(out, v);
  }
}

extern "C" void kernel_launch(void* const* d_in, const int* in_sizes, int n_in,
                              void* d_out, int out_size, void* d_ws, size_t ws_size,
                              hipStream_t stream) {
  const float* X1    = (const float*)d_in[0];
  const float* X2    = (const float*)d_in[1];
  const float* Y     = (const float*)d_in[2];
  const float* w1h_w = (const float*)d_in[3];
  const float* w1h_b = (const float*)d_in[4];
  const float* w2h_w = (const float*)d_in[5];
  const float* w2h_b = (const float*)d_in[6];
  const float* wh1_w = (const float*)d_in[7];
  const float* wh1_b = (const float*)d_in[8];
  const float* wh2_w = (const float*)d_in[9];
  const float* wh2_b = (const float*)d_in[10];
  const float* who_w = (const float*)d_in[11];
  const float* who_b = (const float*)d_in[12];

  // f16 weight copies in workspace (~4 MB)
  _Float16* ws      = (_Float16*)d_ws;
  _Float16* w1h16   = ws;
  _Float16* w2h16   = w1h16  + (size_t)NH * NX;
  _Float16* wh1_16  = w2h16  + (size_t)NH * NX;
  _Float16* wh2_16  = wh1_16 + (size_t)NX * NH;

  zero_f32<<<1, 64, 0, stream>>>((float*)d_out, out_size);

  const int CVT_N = NH * NX;
  cvt_f32_f16<<<(CVT_N + 255) / 256, 256, 0, stream>>>(w1h_w, w1h16, CVT_N);
  cvt_f32_f16<<<(CVT_N + 255) / 256, 256, 0, stream>>>(w2h_w, w2h16, CVT_N);
  cvt_f32_f16<<<(CVT_N + 255) / 256, 256, 0, stream>>>(wh1_w, wh1_16, CVT_N);
  cvt_f32_f16<<<(CVT_N + 255) / 256, 256, 0, stream>>>(wh2_w, wh2_16, CVT_N);

  (void)hipFuncSetAttribute((const void*)fused_mlp_loss,
                            hipFuncAttributeMaxDynamicSharedMemorySize, (int)SMEM_SZ);

  dim3 grid(N_ROWS / M_TILE);   // 2048 blocks x 32 rows
  fused_mlp_loss<<<grid, 256, (size_t)SMEM_SZ, stream>>>(
      X1, X2, Y, w1h_b, w2h_b, wh1_b, wh2_b, who_w, who_b,
      w1h16, w2h16, wh1_16, wh2_16, (float*)d_out);
}
